// SpconvSparseMaskPredictor_26980984553723
// MI455X (gfx1250) — compile-verified, matrix operand order checked by reading, not run on hardware
//
#include <hip/hip_runtime.h>
#include <hip/hip_bf16.h>

// ---------------------------------------------------------------------------
// Types for CDNA5 WMMA (wave32): 16x16x32 f16 -> f32
// ---------------------------------------------------------------------------
typedef __attribute__((ext_vector_type(16))) _Float16 v16h;
typedef __attribute__((ext_vector_type(8)))  _Float16 v8h;
typedef __attribute__((ext_vector_type(8)))  float    v8f;
typedef __attribute__((ext_vector_type(4)))  float    v4f;

static __device__ __forceinline__ v16h mk16(v8h lo, v8h hi) {
    v16h r;
#pragma unroll
    for (int j = 0; j < 8; ++j) { r[j] = lo[j]; r[j + 8] = hi[j]; }
    return r;
}

// Branch-free gelu (tanh approximation) built on raw HW transcendentals:
//   v_exp_f32 (exp2) + v_rcp_f32. ~7 VALU/TRANS ops, no exec-mask divergence.
//   tanh(u) = 1 - 2/(exp(2u)+1);  2*log2(e) = 2.885390082
static __device__ __forceinline__ float gelu_fast(float x) {
    float u = x * fmaf(0.0356774081f * x, x, 0.7978845608f);
    float e = __builtin_amdgcn_exp2f(u * 2.885390082f);      // exp(2u)
    float t = 1.0f - 2.0f * __builtin_amdgcn_rcpf(e + 1.0f); // tanh(u)
    return 0.5f * x * (1.0f + t);
}

// ---------------------------------------------------------------------------
// Pack a fp32 weight matrix W[K][Nc] (row-major, K x Nc) into f16 fragments
// laid out so that for fragment f = nt*kTiles + kk, lane l reads its 16
// B-matrix halves contiguously (two 16-byte loads).
// Lane mapping (16-bit B, 32x16 per WMMA step):
//   n  = nt*16 + (l & 15),  hi = l >> 4
//   slot r<8 : K = kk*32 + hi*8 + r
//   slot r>=8: K = kk*32 + 16 + hi*8 + (r-8)
// ---------------------------------------------------------------------------
__global__ void pack_w_kernel(const float* __restrict__ W, int Ndim,
                              int kTiles, int totalThreads,
                              _Float16* __restrict__ out) {
    int t = blockIdx.x * blockDim.x + threadIdx.x;
    if (t >= totalThreads) return;
    int lane = t & 31;
    int f    = t >> 5;                  // nt*kTiles + kk
    int kk   = f % kTiles;
    int hi   = lane >> 4;
    int nt   = f / kTiles;
    int n    = (nt << 4) + (lane & 15);
    _Float16* dst = out + (size_t)f * 512 + lane * 16;
#pragma unroll
    for (int r = 0; r < 16; ++r) {
        int K = (kk << 5) + ((r < 8) ? ((hi << 3) + r) : (16 + (hi << 3) + (r - 8)));
        dst[r] = (_Float16)W[(size_t)K * Ndim + n];
    }
}

__global__ void zero_sums_kernel(float* __restrict__ gsums, float* __restrict__ gcnt) {
    int t = threadIdx.x;
    if (t < 512) gsums[t] = 0.0f;
    if (t < 8)   gcnt[t]  = 0.0f;
}

// ---------------------------------------------------------------------------
// Stage 1: per-row LayerNorm -> GEMM(128x128, W1) -> gelu.
//   cols 0..63  : store f16 z_local (row-major [N][64])
//   cols 64..127: segment-sum into per-batch accumulators (LDS then global)
// One wave owns 16 rows; block = 256 threads = 8 waves = 128 rows.
// ---------------------------------------------------------------------------
__global__ __launch_bounds__(256) void mlp_stage1_kernel(
    const float* __restrict__ feat, const int* __restrict__ batch_idx,
    const float* __restrict__ gamma, const float* __restrict__ beta,
    const _Float16* __restrict__ w1p, const float* __restrict__ b1,
    _Float16* __restrict__ zloc, float* __restrict__ gsums,
    float* __restrict__ gcnt)
{
    __shared__ float lsums[8 * 64 + 8];
    const int tid = threadIdx.x;
    for (int i = tid; i < 8 * 64 + 8; i += 256) lsums[i] = 0.0f;
    __syncthreads();

    const int lane = tid & 31;
    const int wave = tid >> 5;
    const int hi   = lane >> 4;     // 0: K 0-7/16-23 half, rows 0-7 of C
    const int m    = lane & 15;     // A row / C column within tile
    const long long row0 = (long long)blockIdx.x * 128 + wave * 16;

    // ---- load 64 features of row (row0+m) in A-fragment slot order --------
    const float* rp = feat + (row0 + m) * 128;
    float x[4][16];
#pragma unroll
    for (int kk = 0; kk < 4; ++kk) {
        const int kb = (kk << 5) + (hi << 3);
        v4f q0 = *(const v4f*)(rp + kb);
        v4f q1 = *(const v4f*)(rp + kb + 4);
        v4f q2 = *(const v4f*)(rp + kb + 16);
        v4f q3 = *(const v4f*)(rp + kb + 20);
#pragma unroll
        for (int j = 0; j < 4; ++j) {
            x[kk][j]      = q0[j];
            x[kk][4 + j]  = q1[j];
            x[kk][8 + j]  = q2[j];
            x[kk][12 + j] = q3[j];
        }
    }

    // ---- LayerNorm: lane holds 64 of 128 values, partner lane = l^16 ------
    float s = 0.0f, s2 = 0.0f;
#pragma unroll
    for (int kk = 0; kk < 4; ++kk)
#pragma unroll
        for (int r = 0; r < 16; ++r) { float v = x[kk][r]; s += v; s2 += v * v; }
    s  += __shfl_xor(s, 16);
    s2 += __shfl_xor(s2, 16);
    const float mu  = s * (1.0f / 128.0f);
    const float var = s2 * (1.0f / 128.0f) - mu * mu;
    const float inv = __builtin_amdgcn_rsqf(var + 1e-5f);   // v_rsq_f32

    // ---- normalize, apply gamma/beta, build f16 A fragments ---------------
    v16h a[4];
#pragma unroll
    for (int kk = 0; kk < 4; ++kk) {
        const int kb = (kk << 5) + (hi << 3);
        v4f g0 = *(const v4f*)(gamma + kb);
        v4f g1 = *(const v4f*)(gamma + kb + 4);
        v4f g2 = *(const v4f*)(gamma + kb + 16);
        v4f g3 = *(const v4f*)(gamma + kb + 20);
        v4f e0 = *(const v4f*)(beta + kb);
        v4f e1 = *(const v4f*)(beta + kb + 4);
        v4f e2 = *(const v4f*)(beta + kb + 16);
        v4f e3 = *(const v4f*)(beta + kb + 20);
        v16h t;
#pragma unroll
        for (int j = 0; j < 4; ++j) {
            t[j]      = (_Float16)(((x[kk][j]      - mu) * inv) * g0[j] + e0[j]);
            t[4 + j]  = (_Float16)(((x[kk][4 + j]  - mu) * inv) * g1[j] + e1[j]);
            t[8 + j]  = (_Float16)(((x[kk][8 + j]  - mu) * inv) * g2[j] + e2[j]);
            t[12 + j] = (_Float16)(((x[kk][12 + j] - mu) * inv) * g3[j] + e3[j]);
        }
        a[kk] = t;
    }

    // ---- per-lane batch ids for the 8 C rows this lane holds --------------
    int bids[8];
#pragma unroll
    for (int i = 0; i < 8; ++i) bids[i] = batch_idx[row0 + (hi << 3) + i];

    // ---- GEMM over 8 column tiles of W1 -----------------------------------
#pragma unroll
    for (int nt = 0; nt < 8; ++nt) {
        const float bv = b1[(nt << 4) + m];
        v8f c;
#pragma unroll
        for (int i = 0; i < 8; ++i) c[i] = bv;
#pragma unroll
        for (int kk = 0; kk < 4; ++kk) {
            const _Float16* bp = w1p + (size_t)((nt << 2) + kk) * 512 + lane * 16;
            v8h blo = *(const v8h*)(bp);
            v8h bhi = *(const v8h*)(bp + 8);
            v16h bb = mk16(blo, bhi);
            c = __builtin_amdgcn_wmma_f32_16x16x32_f16(
                    false, a[kk], false, bb, (short)0, c, false, false);
        }
#pragma unroll
        for (int i = 0; i < 8; ++i) c[i] = gelu_fast(c[i]);

        if (nt < 4) {
            // z_local: cols 0..63, row-major f16
#pragma unroll
            for (int i = 0; i < 8; ++i)
                zloc[(row0 + (hi << 3) + i) * 64 + (nt << 4) + m] = (_Float16)c[i];
        } else {
            const int colg = ((nt - 4) << 4) + m;
#pragma unroll
            for (int i = 0; i < 8; ++i)
                atomicAdd(&lsums[bids[i] * 64 + colg], c[i]);   // ds_add_f32
            if (nt == 4 && m == 0) {
#pragma unroll
                for (int i = 0; i < 8; ++i)
                    atomicAdd(&lsums[512 + bids[i]], 1.0f);
            }
        }
    }

    __syncthreads();
    for (int i = tid; i < 512; i += 256) atomicAdd(&gsums[i], lsums[i]);
    if (tid < 8) atomicAdd(&gcnt[tid], lsums[512 + tid]);
}

// ---------------------------------------------------------------------------
// Stage 1.5: means[b][c] = sums[b][c] / max(count[b],1), stored as f16.
// ---------------------------------------------------------------------------
__global__ void means_kernel(const float* __restrict__ gsums,
                             const float* __restrict__ gcnt,
                             _Float16* __restrict__ means_h) {
    int t = threadIdx.x;
    if (t < 512) {
        int b = t >> 6;
        means_h[t] = (_Float16)(gsums[t] / fmaxf(gcnt[b], 1.0f));
    }
}

// ---------------------------------------------------------------------------
// Stage 2: x2 = [z_local | means[batch]] (16x128 f16 per wave-tile)
//   h2 = gelu(x2 @ W2 + b2)  (16x64)  -> transpose via LDS
//   h3 = gelu(h2 @ W3 + b3)  (16x32)
//   out = h3 @ W4 + b4       (16x1)   via shuffle reduction
// ---------------------------------------------------------------------------
__global__ __launch_bounds__(256) void mlp_stage2_kernel(
    const _Float16* __restrict__ zloc, const int* __restrict__ batch_idx,
    const _Float16* __restrict__ means_h,
    const _Float16* __restrict__ w2p, const float* __restrict__ b2,
    const _Float16* __restrict__ w3p, const float* __restrict__ b3,
    const float* __restrict__ w4, const float* __restrict__ b4,
    float* __restrict__ out)
{
    __shared__ __align__(16) _Float16 h2buf[8][16][64];
    const int tid  = threadIdx.x;
    const int lane = tid & 31;
    const int wave = tid >> 5;
    const int hi   = lane >> 4;
    const int m    = lane & 15;
    const long long row0 = (long long)blockIdx.x * 128 + wave * 16;

    // ---- A fragments: kk 0..1 from z_local, kk 2..3 from per-batch means --
    v16h a[4];
    const _Float16* zr = zloc + (row0 + m) * 64;
#pragma unroll
    for (int kk = 0; kk < 2; ++kk) {
        v8h g0 = *(const v8h*)(zr + (kk << 5) + (hi << 3));
        v8h g1 = *(const v8h*)(zr + (kk << 5) + 16 + (hi << 3));
        a[kk] = mk16(g0, g1);
    }
    const int bm = batch_idx[row0 + m];
    const _Float16* mr = means_h + bm * 64;
#pragma unroll
    for (int kk = 2; kk < 4; ++kk) {
        const int kb = ((kk - 2) << 5) + (hi << 3);
        v8h g0 = *(const v8h*)(mr + kb);
        v8h g1 = *(const v8h*)(mr + kb + 16);
        a[kk] = mk16(g0, g1);
    }

    // ---- GEMM W2: 4 column tiles, K=128 -----------------------------------
#pragma unroll
    for (int nt = 0; nt < 4; ++nt) {
        const float bv = b2[(nt << 4) + m];
        v8f c;
#pragma unroll
        for (int i = 0; i < 8; ++i) c[i] = bv;
#pragma unroll
        for (int kk = 0; kk < 4; ++kk) {
            const _Float16* bp = w2p + (size_t)((nt << 2) + kk) * 512 + lane * 16;
            v16h bb = mk16(*(const v8h*)bp, *(const v8h*)(bp + 8));
            c = __builtin_amdgcn_wmma_f32_16x16x32_f16(
                    false, a[kk], false, bb, (short)0, c, false, false);
        }
#pragma unroll
        for (int i = 0; i < 8; ++i)
            h2buf[wave][(hi << 3) + i][(nt << 4) + m] = (_Float16)gelu_fast(c[i]);
    }
    __syncthreads();

    // ---- A fragments for W3 from transposed h2 ----------------------------
    v16h a2[2];
    const _Float16* hr = &h2buf[wave][m][0];
#pragma unroll
    for (int kk = 0; kk < 2; ++kk) {
        v8h g0 = *(const v8h*)(hr + (kk << 5) + (hi << 3));       // ds_load_b128
        v8h g1 = *(const v8h*)(hr + (kk << 5) + 16 + (hi << 3));
        a2[kk] = mk16(g0, g1);
    }

    // ---- GEMM W3: 2 column tiles, K=64 ------------------------------------
    v8f c3[2];
#pragma unroll
    for (int nt = 0; nt < 2; ++nt) {
        const float bv = b3[(nt << 4) + m];
        v8f c;
#pragma unroll
        for (int i = 0; i < 8; ++i) c[i] = bv;
#pragma unroll
        for (int kk = 0; kk < 2; ++kk) {
            const _Float16* bp = w3p + (size_t)((nt << 1) + kk) * 512 + lane * 16;
            v16h bb = mk16(*(const v8h*)bp, *(const v8h*)(bp + 8));
            c = __builtin_amdgcn_wmma_f32_16x16x32_f16(
                    false, a2[kk], false, bb, (short)0, c, false, false);
        }
#pragma unroll
        for (int i = 0; i < 8; ++i) c[i] = gelu_fast(c[i]);
        c3[nt] = c;
    }

    // ---- Final 32 -> 1 projection: per-row dot via shuffle butterfly ------
    const float wa = w4[m];
    const float wb = w4[16 + m];
    const float bias4 = b4[0];
    float res[8];
#pragma unroll
    for (int i = 0; i < 8; ++i) {
        float p = c3[0][i] * wa + c3[1][i] * wb;
        p += __shfl_xor(p, 1);
        p += __shfl_xor(p, 2);
        p += __shfl_xor(p, 4);
        p += __shfl_xor(p, 8);
        res[i] = p + bias4;
    }
    if (m == 0) {
#pragma unroll
        for (int i = 0; i < 8; ++i)
            out[row0 + (hi << 3) + i] = res[i];
    }
}

// ---------------------------------------------------------------------------
// Launcher
// ---------------------------------------------------------------------------
extern "C" void kernel_launch(void* const* d_in, const int* in_sizes, int n_in,
                              void* d_out, int out_size, void* d_ws, size_t ws_size,
                              hipStream_t stream) {
    const float* feat  = (const float*)d_in[0];
    const int*   bidx  = (const int*)d_in[1];
    const float* gamma = (const float*)d_in[2];
    const float* beta  = (const float*)d_in[3];
    const float* W1    = (const float*)d_in[4];
    const float* b1    = (const float*)d_in[5];
    const float* W2    = (const float*)d_in[6];
    const float* b2    = (const float*)d_in[7];
    const float* W3    = (const float*)d_in[8];
    const float* b3    = (const float*)d_in[9];
    const float* W4    = (const float*)d_in[10];
    const float* b4    = (const float*)d_in[11];
    float* outp = (float*)d_out;

    const long long N = in_sizes[1];   // 524288 rows (batch_idx length)

    char* ws = (char*)d_ws;
    _Float16* w1p     = (_Float16*)(ws);             // 8*4*512*2   = 32768 B
    _Float16* w2p     = (_Float16*)(ws + 32768);     // 4*4*512*2   = 16384 B
    _Float16* w3p     = (_Float16*)(ws + 49152);     // 2*2*512*2   =  4096 B
    float*    gsums   = (float*)   (ws + 53248);     // 8*64*4      =  2048 B
    float*    gcnt    = (float*)   (ws + 55296);     //               32 B
    _Float16* means_h = (_Float16*)(ws + 55424);     // 8*64*2      =  1024 B
    _Float16* zloc    = (_Float16*)(ws + 65536);     // N*64*2      =   64 MB

    // Pack weights into WMMA-B fragment layout (f16).
    pack_w_kernel<<<(8 * 4 * 32 + 255) / 256, 256, 0, stream>>>(W1, 128, 4, 8 * 4 * 32, w1p);
    pack_w_kernel<<<(4 * 4 * 32 + 255) / 256, 256, 0, stream>>>(W2,  64, 4, 4 * 4 * 32, w2p);
    pack_w_kernel<<<1, 128, 0, stream>>>(W3, 32, 2, 2 * 2 * 32, w3p);

    zero_sums_kernel<<<1, 512, 0, stream>>>(gsums, gcnt);

    const int blocks = (int)(N / 128);   // 4096
    mlp_stage1_kernel<<<blocks, 256, 0, stream>>>(feat, bidx, gamma, beta,
                                                  w1p, b1, zloc, gsums, gcnt);
    means_kernel<<<1, 512, 0, stream>>>(gsums, gcnt, means_h);
    mlp_stage2_kernel<<<blocks, 256, 0, stream>>>(zloc, bidx, means_h,
                                                  w2p, b2, w3p, b3, W4, b4, outp);
}
